// STMGAT_4518305595894
// MI455X (gfx1250) — compile-verified, matrix operand order checked by reading, not run on hardware
//
#include <hip/hip_runtime.h>
#include <hip/hip_bf16.h>

// ---- model constants (match reference) ----
#define NN    207
#define BB    16
#define TT    12
#define RFP   13          // receptive field (padded time length)
#define CRES  32
#define CSKIP 256
#define CEND  512
#define HEADS 4
#define OWID  12
#define LTOT  8

typedef __attribute__((ext_vector_type(16))) _Float16 v16h;
typedef __attribute__((ext_vector_type(8)))  _Float16 v8h;
typedef __attribute__((ext_vector_type(8)))  float    v8f;

static __device__ __forceinline__ float lrelu(float x, float s){ return x > 0.f ? x : s*x; }

// ---------------- generic converts ----------------
__global__ void k_cvt(const float* __restrict__ src, _Float16* __restrict__ dst, int n){
  int i = blockIdx.x*blockDim.x + threadIdx.x;
  if (i < n) dst[i] = (_Float16)src[i];
}
// src[rows,cols] -> dst[rowsPad,cols], pad rows zero-filled
__global__ void k_cvt_pad(const float* __restrict__ src, _Float16* __restrict__ dst,
                          int rows, int cols, int rowsPad){
  int i = blockIdx.x*blockDim.x + threadIdx.x;
  if (i >= rowsPad*cols) return;
  int r = i / cols;
  dst[i] = (r < rows) ? (_Float16)src[i] : (_Float16)0;
}
__global__ void k_relu_cvt_pad(const float* __restrict__ src, _Float16* __restrict__ dst,
                               int rows, int cols, int rowsPad){
  int i = blockIdx.x*blockDim.x + threadIdx.x;
  if (i >= rowsPad*cols) return;
  int r = i / cols;
  float v = (r < rows) ? src[i] : 0.f;
  dst[i] = (_Float16)(v > 0.f ? v : 0.f);
}
// src[R,C] row-major -> dst[C,R] row-major (transpose + f16); C assumed %32==0
__global__ void k_cvt_t(const float* __restrict__ src, _Float16* __restrict__ dst, int R, int C){
  int i = blockIdx.x*blockDim.x + threadIdx.x;
  if (i < R*C){ int r = i / C, c = i - r*C; dst[c*R + r] = (_Float16)src[i]; }
}
__global__ void k_zero(float* __restrict__ p, int n){
  int i = blockIdx.x*blockDim.x + threadIdx.x;
  if (i < n) p[i] = 0.f;
}

// ---------------- WMMA GEMM ----------------
// C[M,N] = A[Mpad,K] (f16 row-major, rows >= M zeroed) * Bt[Npad,K]^T (f16, cols >= N zeroed)
// K % 32 == 0. One wave -> 16x32 of C (two 16x16 tiles, shared A fragment).
// grid.x = Npad/32, grid.y = Mpad/64, block = 128 (4 waves).
__global__ void k_gemm(const _Float16* __restrict__ A, const _Float16* __restrict__ Bt,
                       float* __restrict__ C, int M, int N, int K,
                       const float* __restrict__ bias, int act){
  int lane = threadIdx.x & 31;
  int wave = threadIdx.x >> 5;
  int m0 = (blockIdx.y*4 + wave) * 16;
  int n0 = blockIdx.x * 32;
  int l15 = lane & 15;
  int hlf = lane >> 4;
  // lane's fragment = two contiguous 16B chunks: K in [hlf*8, hlf*8+7] and [16+hlf*8, ...]
  const _Float16* arow  = A  + (size_t)(m0 + l15) * K + (hlf << 3);
  const _Float16* bcol0 = Bt + (size_t)(n0 + l15) * K + (hlf << 3);
  const _Float16* bcol1 = bcol0 + (size_t)16 * K;
  v8f acc0 = {}, acc1 = {};
#pragma unroll 2
  for (int k0 = 0; k0 < K; k0 += 32){
    __builtin_prefetch(arow  + k0 + 64, 0, 1);
    __builtin_prefetch(bcol0 + k0 + 64, 0, 1);
    v8h a_lo = *(const v8h*)(arow  + k0);
    v8h a_hi = *(const v8h*)(arow  + k0 + 16);
    v8h b0lo = *(const v8h*)(bcol0 + k0);
    v8h b0hi = *(const v8h*)(bcol0 + k0 + 16);
    v8h b1lo = *(const v8h*)(bcol1 + k0);
    v8h b1hi = *(const v8h*)(bcol1 + k0 + 16);
    v16h a  = __builtin_shufflevector(a_lo, a_hi, 0,1,2,3,4,5,6,7,8,9,10,11,12,13,14,15);
    v16h b0 = __builtin_shufflevector(b0lo, b0hi, 0,1,2,3,4,5,6,7,8,9,10,11,12,13,14,15);
    v16h b1 = __builtin_shufflevector(b1lo, b1hi, 0,1,2,3,4,5,6,7,8,9,10,11,12,13,14,15);
    acc0 = __builtin_amdgcn_wmma_f32_16x16x32_f16(false, a, false, b0, (short)0, acc0,
                                                  false, false);
    acc1 = __builtin_amdgcn_wmma_f32_16x16x32_f16(false, a, false, b1, (short)0, acc1,
                                                  false, false);
  }
  int mBase = m0 + (hlf << 3);
#pragma unroll
  for (int r = 0; r < 8; ++r){
    int m = mBase + r;
    if (m >= M) continue;
    int n = n0 + l15;
    if (n < N){
      float v = acc0[r];
      if (bias) v += bias[n];
      if (act == 1) v = fmaxf(v, 0.f);
      C[(size_t)m*N + n] = v;
    }
    int n2 = n + 16;
    if (n2 < N){
      float v = acc1[r];
      if (bias) v += bias[n2];
      if (act == 1) v = fmaxf(v, 0.f);
      C[(size_t)m*N + n2] = v;
    }
  }
}

// ---------------- front end ----------------
__global__ void k_start(const float* __restrict__ inp, const float* __restrict__ tim,
                        const float* __restrict__ sw, const float* __restrict__ sb,
                        const float* __restrict__ cw, const float* __restrict__ cb,
                        float* __restrict__ x){
  int i = blockIdx.x*blockDim.x + threadIdx.x;
  if (i >= BB*CRES*NN*RFP) return;
  int t = i % RFP; int n = (i/RFP) % NN; int c = (i/(RFP*NN)) % CRES; int b = i/(RFP*NN*CRES);
  float in0 = 0.f, in1 = 0.f;
  if (t > 0){
    in0 = inp[(b*TT + (t-1))*NN + n];
    in1 = tim[b*TT + (t-1)];
  }
  float s  = sw[c*2+0]*in0 + sw[c*2+1]*in1 + sb[c];
  float cc = cw[c*2+0]*in0 + cw[c*2+1]*in1 + cb[c];
  x[i] = s + lrelu(cc, 0.01f);
}

// ---------------- dilated conv as GEMM ----------------
// A[row*64 + tap*32 + c] = x[b,c,n,t+tap*D], row = (b*N+n)*tout+t ; rows >= M zeroed
__global__ void k_dconv_a(const float* __restrict__ x, _Float16* __restrict__ A,
                          int tin, int tout, int D, int M, int Mpad){
  int i = blockIdx.x*blockDim.x + threadIdx.x;
  if (i >= Mpad*64) return;
  int row = i >> 6;
  if (row >= M){ A[i] = (_Float16)0; return; }
  int c = i & 31; int tap = (i >> 5) & 1;
  int t = row % tout; int n = (row/tout) % NN; int b = row/(tout*NN);
  A[i] = (_Float16)x[((b*CRES + c)*NN + n)*tin + t + tap*D];
}
__global__ void k_dconv_w(const float* __restrict__ fw, const float* __restrict__ gw,
                          _Float16* __restrict__ Bt){
  int i = blockIdx.x*blockDim.x + threadIdx.x;
  if (i >= 64*64) return;
  int c = i & 31; int tap = (i >> 5) & 1; int o2 = i >> 6;
  const float* w = (o2 < 32) ? fw : gw;
  int o = o2 & 31;
  Bt[i] = (_Float16)w[(o*CRES + c)*2 + tap];
}
__global__ void k_gateact(const float* __restrict__ Y, const float* __restrict__ fb,
                          const float* __restrict__ gb, float* __restrict__ xg, int tout){
  int i = blockIdx.x*blockDim.x + threadIdx.x;
  int M = BB*NN*tout;
  if (i >= M*CRES) return;
  int c = i & 31; int row = i >> 5;
  int t = row % tout; int n = (row/tout) % NN; int b = row/(tout*NN);
  float f = tanhf(Y[(size_t)row*64 + c] + fb[c]);
  float g = Y[(size_t)row*64 + 32 + c] + gb[c];
  g = 1.f/(1.f + expf(-g));
  xg[((b*CRES + c)*NN + n)*tout + t] = f*g;
}

// ---------------- skip path ----------------
__global__ void k_lastcol(const float* __restrict__ x, _Float16* __restrict__ A,
                          int tlen, int Mpad){
  int i = blockIdx.x*blockDim.x + threadIdx.x;   // (b*N+n)*32 + c
  if (i >= Mpad*CRES) return;
  int c = i & 31; int bn = i >> 5;
  if (bn >= BB*NN){ A[i] = (_Float16)0; return; }
  int n = bn % NN; int b = bn / NN;
  A[i] = (_Float16)x[((b*CRES + c)*NN + n)*tlen + (tlen - 1)];
}
__global__ void k_skipacc(const float* __restrict__ Y, const float* __restrict__ sb,
                          float* __restrict__ skip){
  int i = blockIdx.x*blockDim.x + threadIdx.x;
  if (i >= BB*NN*CSKIP) return;
  int o = i & 255;
  skip[i] += Y[i] + sb[o];
}

// ---------------- CSR of incoming edges (deterministic) ----------------
__global__ void k_csr_count(const int* __restrict__ dst, int* __restrict__ counts, int E){
  int n = blockIdx.x*blockDim.x + threadIdx.x;
  if (n >= NN) return;
  int c = 0;
  for (int e = 0; e < E; ++e) c += (dst[e] == n);
  counts[n] = c;
}
__global__ void k_csr_scan(const int* __restrict__ counts, int* __restrict__ rowptr){
  if (blockIdx.x || threadIdx.x) return;
  int acc = 0; rowptr[0] = 0;
  for (int n = 0; n < NN; ++n){ acc += counts[n]; rowptr[n+1] = acc; }
}
__global__ void k_csr_fill(const int* __restrict__ dst, const int* __restrict__ rowptr,
                           int* __restrict__ eidx, int E){
  int n = blockIdx.x*blockDim.x + threadIdx.x;
  if (n >= NN) return;
  int p = rowptr[n];
  for (int e = 0; e < E; ++e) if (dst[e] == n) eidx[p++] = e;
}

// ---------------- GAT pieces ----------------
// A[(b*N+n)*Fi + c*tout + t] = xg[b,c,n,t]; rows >= B*N zeroed
__global__ void k_h16(const float* __restrict__ xg, _Float16* __restrict__ A,
                      int tout, int Mpad){
  int Fi = CRES*tout;
  int i = blockIdx.x*blockDim.x + threadIdx.x;
  if (i >= Mpad*Fi) return;
  int f = i % Fi; int bn = i / Fi;
  if (bn >= BB*NN){ A[i] = (_Float16)0; return; }
  int c = f / tout; int t = f - c*tout;
  int n = bn % NN; int b = bn / NN;
  A[i] = (_Float16)xg[((b*CRES + c)*NN + n)*tout + t];
}
__global__ void k_elr(const float* __restrict__ z, const float* __restrict__ al,
                      const float* __restrict__ ar, float* __restrict__ el,
                      float* __restrict__ er, int Fi){
  int i = blockIdx.x*blockDim.x + threadIdx.x;   // (b*N+n)*H + h
  if (i >= BB*NN*HEADS) return;
  int h = i & 3; int bn = i >> 2;
  const float* zp  = z + ((size_t)bn*HEADS + h)*Fi;
  const float* alp = al + h*Fi;
  const float* arp = ar + h*Fi;
  float sl = 0.f, sr = 0.f;
  for (int f = 0; f < Fi; ++f){ float zv = zp[f]; sl += zv*alp[f]; sr += zv*arp[f]; }
  el[i] = sl; er[i] = sr;
}
__global__ void k_edge(const float* __restrict__ el, const float* __restrict__ er,
                       const int* __restrict__ src, const int* __restrict__ dst,
                       float* __restrict__ esc, int E){
  int i = blockIdx.x*blockDim.x + threadIdx.x;   // ((b*E)+e)*H + h
  if (i >= BB*E*HEADS) return;
  int h = i & 3; int be = i >> 2; int e = be % E; int b = be / E;
  float v = el[(b*NN + src[e])*HEADS + h] + er[(b*NN + dst[e])*HEADS + h];
  esc[i] = lrelu(v, 0.2f);
}
__global__ void k_segmax(const float* __restrict__ esc, const int* __restrict__ rowptr,
                         const int* __restrict__ eidx, float* __restrict__ nmax, int E){
  int i = blockIdx.x*blockDim.x + threadIdx.x;   // (b*N+n)*H + h
  if (i >= BB*NN*HEADS) return;
  int h = i & 3; int bn = i >> 2; int n = bn % NN; int b = bn / NN;
  float m = -3.0e38f;
  for (int p = rowptr[n]; p < rowptr[n+1]; ++p){
    float v = esc[((size_t)b*E + eidx[p])*HEADS + h];
    m = fmaxf(m, v);
  }
  nmax[i] = m;
}
__global__ void k_expsum(float* __restrict__ esc, const int* __restrict__ rowptr,
                         const int* __restrict__ eidx, const float* __restrict__ nmax,
                         float* __restrict__ nsum, int E){
  int i = blockIdx.x*blockDim.x + threadIdx.x;
  if (i >= BB*NN*HEADS) return;
  int h = i & 3; int bn = i >> 2; int n = bn % NN; int b = bn / NN;
  float m = nmax[i], s = 0.f;
  for (int p = rowptr[n]; p < rowptr[n+1]; ++p){
    size_t idx = ((size_t)b*E + eidx[p])*HEADS + h;
    float ex = expf(esc[idx] - m);
    esc[idx] = ex;
    s += ex;
  }
  nsum[i] = s;
}
__global__ void k_agg(const float* __restrict__ z, const float* __restrict__ esc,
                      const float* __restrict__ nsum, const int* __restrict__ rowptr,
                      const int* __restrict__ eidx, const int* __restrict__ src,
                      const float* __restrict__ bias, float* __restrict__ hout,
                      int Fi, int E){
  int i = blockIdx.x*blockDim.x + threadIdx.x;   // (b*N+n)*Fi + f
  if (i >= BB*NN*Fi) return;
  int f = i % Fi; int bn = i / Fi; int n = bn % NN; int b = bn / NN;
  int p0 = rowptr[n], p1 = rowptr[n+1];
  float acc = 0.f;
  for (int h = 0; h < HEADS; ++h){
    float den = nsum[bn*HEADS + h];
    float s = 0.f;
    for (int p = p0; p < p1; ++p){
      int e = eidx[p];
      float a = esc[((size_t)b*E + e)*HEADS + h];
      s += a * z[((size_t)(b*NN + src[e])*HEADS + h)*Fi + f];
    }
    s = s/den + bias[h*Fi + f];
    acc += (s > 0.f) ? s : (expf(s) - 1.f);
  }
  hout[i] = acc * 0.25f;
}

// ---------------- residual combine + batchnorm ----------------
__global__ void k_combine(const float* __restrict__ xg, const float* __restrict__ h2,
                          const float* __restrict__ xres, float* __restrict__ xo,
                          int tin, int tout, int D){
  int i = blockIdx.x*blockDim.x + threadIdx.x;   // [B,32,N,tout]
  if (i >= BB*CRES*NN*tout) return;
  int t = i % tout; int n = (i/tout) % NN; int c = (i/(tout*NN)) & 31; int b = i/(tout*NN*CRES);
  int Fi = CRES*tout;
  xo[i] = xg[i] + h2[(size_t)(b*NN + n)*Fi + c*tout + t]
               + xres[((b*CRES + c)*NN + n)*tin + t + D];
}
__global__ void k_bnstat(const float* __restrict__ x, float* __restrict__ mean,
                         float* __restrict__ inv, int tout){
  int c = blockIdx.x;                // 32 blocks, one per channel
  int cnt = BB*NN*tout;
  float s = 0.f, s2 = 0.f;
  for (int j = threadIdx.x; j < cnt; j += blockDim.x){
    int t = j % tout; int n = (j/tout) % NN; int b = j/(tout*NN);
    float v = x[((b*CRES + c)*NN + n)*tout + t];
    s += v; s2 += v*v;
  }
  __shared__ float sh[256], sh2[256];
  sh[threadIdx.x] = s; sh2[threadIdx.x] = s2;
  __syncthreads();
  for (int o = blockDim.x >> 1; o > 0; o >>= 1){
    if ((int)threadIdx.x < o){ sh[threadIdx.x] += sh[threadIdx.x+o]; sh2[threadIdx.x] += sh2[threadIdx.x+o]; }
    __syncthreads();
  }
  if (threadIdx.x == 0){
    float m = sh[0]/cnt;
    float var = sh2[0]/cnt - m*m;
    mean[c] = m; inv[c] = rsqrtf(var + 1e-5f);
  }
}
__global__ void k_bnapply(float* __restrict__ x, const float* __restrict__ mean,
                          const float* __restrict__ inv, const float* __restrict__ g,
                          const float* __restrict__ bta, int tout){
  int i = blockIdx.x*blockDim.x + threadIdx.x;
  if (i >= BB*CRES*NN*tout) return;
  int c = (i/(tout*NN)) & 31;
  x[i] = (x[i] - mean[c])*inv[c]*g[c] + bta[c];
}

// ---------------- final output transpose ----------------
__global__ void k_finout(const float* __restrict__ Y, float* __restrict__ out){
  int i = blockIdx.x*blockDim.x + threadIdx.x;   // out[b,ow,n]
  if (i >= BB*OWID*NN) return;
  int n = i % NN; int ow = (i/NN) % OWID; int b = i/(NN*OWID);
  out[i] = Y[(size_t)(b*NN + n)*OWID + ow];
}

// ============================================================
extern "C" void kernel_launch(void* const* d_in, const int* in_sizes, int n_in,
                              void* d_out, int out_size, void* d_ws, size_t ws_size,
                              hipStream_t stream){
  (void)n_in; (void)out_size; (void)ws_size;
  const float* inp   = (const float*)d_in[0];
  const float* tim   = (const float*)d_in[1];
  const int*   src   = (const int*)d_in[2];
  const int*   dst   = (const int*)d_in[3];
  const float* sw    = (const float*)d_in[4];
  const float* sb    = (const float*)d_in[5];
  const float* cw    = (const float*)d_in[6];
  const float* cb    = (const float*)d_in[7];
  const float* fwAll = (const float*)d_in[8];
  const float* fbAll = (const float*)d_in[9];
  const float* gwAll = (const float*)d_in[10];
  const float* gbAll = (const float*)d_in[11];
  const float* swAll = (const float*)d_in[12];
  const float* sbAll = (const float*)d_in[13];
  const float* bngAll= (const float*)d_in[14];
  const float* bnbAll= (const float*)d_in[15];
  const int GP = 16;                       // gat params: 7 layers x 8 arrays
  const float* e1w = (const float*)d_in[GP + 7*8 + 0];
  const float* e1b = (const float*)d_in[GP + 7*8 + 1];
  const float* e2w = (const float*)d_in[GP + 7*8 + 2];
  const float* e2b = (const float*)d_in[GP + 7*8 + 3];
  const int E = in_sizes[2];

  // ---- carve workspace ----
  char* base = (char*)d_ws; size_t off = 0;
  auto alloc = [&](size_t bytes)->void*{
    void* p = base + off; off = (off + bytes + 255) & ~(size_t)255; return p;
  };
  const size_t XBUF = (size_t)BB*CRES*NN*RFP;
  const int MBN  = BB*NN;                       // 3312
  const int MBNp = ((MBN + 63)/64)*64;          // 3328
  float* xA   = (float*)alloc(XBUF*4);
  float* xB   = (float*)alloc(XBUF*4);
  float* xG   = (float*)alloc(XBUF*4);
  float* skip = (float*)alloc((size_t)MBN*CSKIP*4);
  float* zbuf = (float*)alloc((size_t)MBN*HEADS*384*4);
  float* el   = (float*)alloc((size_t)MBN*HEADS*4);
  float* er   = (float*)alloc((size_t)MBN*HEADS*4);
  float* nmax = (float*)alloc((size_t)MBN*HEADS*4);
  float* nsum = (float*)alloc((size_t)MBN*HEADS*4);
  float* esc  = (float*)alloc((size_t)BB*E*HEADS*4);
  float* h1   = (float*)alloc((size_t)MBN*384*4);
  float* h2   = (float*)alloc((size_t)MBN*384*4);
  float* bnm  = (float*)alloc(32*4);
  float* bni  = (float*)alloc(32*4);
  int* counts = (int*)alloc(NN*4);
  int* rowptr = (int*)alloc((NN+1)*4);
  int* eidx   = (int*)alloc((size_t)E*4);
  _Float16* A16 = (_Float16*)alloc((size_t)(BB*NN*TT + 64)*64*2); // >= all padded A
  _Float16* B16 = (_Float16*)alloc((size_t)384*4*384*2);          // >= all padded Bt

  auto G = [](long n, int b){ return dim3((unsigned)((n + b - 1)/b)); };
  const int TB = 256;

  // CSR of incoming edges (deterministic)
  k_csr_count<<<G(NN,64), 64, 0, stream>>>(dst, counts, E);
  k_csr_scan <<<1, 1, 0, stream>>>(counts, rowptr);
  k_csr_fill <<<G(NN,64), 64, 0, stream>>>(dst, rowptr, eidx, E);

  k_zero<<<G((long)MBN*CSKIP,TB), TB, 0, stream>>>(skip, MBN*CSKIP);
  k_start<<<G((long)BB*CRES*NN*RFP,TB), TB, 0, stream>>>(inp, tim, sw, sb, cw, cb, xA);

  const int DILS[LTOT] = {1,2,1,2,1,2,1,2};
  int tin = RFP;
  float* xcur = xA;
  for (int i = 0; i < LTOT; ++i){
    int D = DILS[i], tout = tin - D, Fi = CRES*tout;
    int M = MBN*tout;
    int Mpad = ((M + 63)/64)*64;
    // dilated conv (filter||gate) as WMMA GEMM  [M,64]x[64,64]
    k_dconv_a<<<G((long)Mpad*64,TB), TB, 0, stream>>>(xcur, A16, tin, tout, D, M, Mpad);
    k_dconv_w<<<G(64*64,TB), TB, 0, stream>>>(fwAll + (size_t)i*CRES*CRES*2,
                                              gwAll + (size_t)i*CRES*CRES*2, B16);
    { dim3 g(64/32, (unsigned)(Mpad/64));
      k_gemm<<<g, 128, 0, stream>>>(A16, B16, zbuf, M, 64, 64, nullptr, 0); }
    k_gateact<<<G((long)M*CRES,TB), TB, 0, stream>>>(zbuf, fbAll + i*CRES, gbAll + i*CRES,
                                                     xG, tout);

    // skip conv on last time column  [B*N,32]x[32,256]
    k_lastcol<<<G((long)MBNp*CRES,TB), TB, 0, stream>>>(xG, A16, tout, MBNp);
    k_cvt<<<G(CSKIP*CRES,TB), TB, 0, stream>>>(swAll + (size_t)i*CSKIP*CRES, B16, CSKIP*CRES);
    { dim3 g(CSKIP/32, (unsigned)(MBNp/64));
      k_gemm<<<g, 128, 0, stream>>>(A16, B16, zbuf, MBN, CSKIP, CRES, nullptr, 0); }
    k_skipacc<<<G((long)MBN*CSKIP,TB), TB, 0, stream>>>(zbuf, sbAll + i*CSKIP, skip);

    if (i == LTOT - 1) break;

    // ---- two GAT layers ----
    const float* p[8];
    for (int q = 0; q < 8; ++q) p[q] = (const float*)d_in[GP + i*8 + q];
    for (int gpass = 0; gpass < 2; ++gpass){
      const float* W  = p[gpass*4 + 0];
      const float* al = p[gpass*4 + 1];
      const float* ar = p[gpass*4 + 2];
      const float* bi = p[gpass*4 + 3];
      if (gpass == 0)
        k_h16<<<G((long)MBNp*Fi,TB), TB, 0, stream>>>(xG, A16, tout, MBNp);
      else
        k_cvt_pad<<<G((long)MBNp*Fi,TB), TB, 0, stream>>>(h1, A16, MBN, Fi, MBNp);
      k_cvt_t<<<G((long)Fi*HEADS*Fi,TB), TB, 0, stream>>>(W, B16, Fi, HEADS*Fi);
      { dim3 g((HEADS*Fi)/32, (unsigned)(MBNp/64));
        k_gemm<<<g, 128, 0, stream>>>(A16, B16, zbuf, MBN, HEADS*Fi, Fi, nullptr, 0); }
      k_elr   <<<G((long)MBN*HEADS,TB), TB, 0, stream>>>(zbuf, al, ar, el, er, Fi);
      k_edge  <<<G((long)BB*E*HEADS,TB), TB, 0, stream>>>(el, er, src, dst, esc, E);
      k_segmax<<<G((long)MBN*HEADS,TB), TB, 0, stream>>>(esc, rowptr, eidx, nmax, E);
      k_expsum<<<G((long)MBN*HEADS,TB), TB, 0, stream>>>(esc, rowptr, eidx, nmax, nsum, E);
      k_agg   <<<G((long)MBN*Fi,TB), TB, 0, stream>>>(zbuf, esc, nsum, rowptr, eidx, src,
                                                      bi, gpass == 0 ? h1 : h2, Fi, E);
    }

    // combine + batchnorm
    float* xnext = (xcur == xA) ? xB : xA;
    k_combine<<<G((long)BB*CRES*NN*tout,TB), TB, 0, stream>>>(xG, h2, xcur, xnext, tin, tout, D);
    k_bnstat <<<32, 256, 0, stream>>>(xnext, bnm, bni, tout);
    k_bnapply<<<G((long)BB*CRES*NN*tout,TB), TB, 0, stream>>>(xnext, bnm, bni,
                                                              bngAll + i*CRES, bnbAll + i*CRES,
                                                              tout);
    xcur = xnext;
    tin = tout;
  }

  // ---- end head: relu(skip) -> end1 (relu) -> end2 ----
  k_relu_cvt_pad<<<G((long)MBNp*CSKIP,TB), TB, 0, stream>>>(skip, A16, MBN, CSKIP, MBNp);
  k_cvt<<<G((long)CEND*CSKIP,TB), TB, 0, stream>>>(e1w, B16, CEND*CSKIP);
  { dim3 g(CEND/32, (unsigned)(MBNp/64));
    k_gemm<<<g, 128, 0, stream>>>(A16, B16, zbuf, MBN, CEND, CSKIP, e1b, 1); }
  k_cvt_pad<<<G((long)MBNp*CEND,TB), TB, 0, stream>>>(zbuf, A16, MBN, CEND, MBNp);
  k_cvt_pad<<<G((long)32*CEND,TB), TB, 0, stream>>>(e2w, B16, OWID, CEND, 32);
  { dim3 g(1, (unsigned)(MBNp/64));
    k_gemm<<<g, 128, 0, stream>>>(A16, B16, zbuf, MBN, OWID, CEND, e2b, 0); }
  k_finout<<<G((long)BB*OWID*NN,TB), TB, 0, stream>>>(zbuf, (float*)d_out);
}
